// GroupRigidParamVelocityStrategy_43860206026939
// MI455X (gfx1250) — matrix operand order repaired and checked by reading, not compiled, not run on hardware
//
#include <hip/hip_runtime.h>
#include <hip/hip_bf16.h>
#include <math.h>

// ---------------------------------------------------------------------------
// GroupRigidParamVelocityStrategy, CDNA5 (gfx1250, wave32, WMMA)
//
// Pipeline:
//  K0 prep  : W1 (512x256 f32) -> W1t (256x512 bf16, N-major)
//             W2 (256x9  f32) -> W2t (16x512? no: 16x256 bf16, N-major, cols>=9 zero)
//  K1 fused : per 16 group-rows: mean(4 token rows)->bf16 LDS A-tile,
//             GEMM1 (16x512 @ 512x256) via v_wmma_f32_16x16x32_bf16,
//             +b1, silu -> bf16 LDS H-tile,
//             GEMM2 (16x256 @ 256x16pad) via WMMA (wave 0),
//             +b2 -> params, 6D rotation, rigid transform, v=(x_goal-kp)/tau
//  K2 tang  : per batch: chain-edge tridiagonal (Thomas) solve, out = v - v_corr
// ---------------------------------------------------------------------------

typedef __attribute__((ext_vector_type(16))) __bf16 v16bf;
typedef __attribute__((ext_vector_type(8)))  float  v8f;

union FragAB { v16bf v; __bf16 e[16]; uint4 q[2]; };
union FragC  { v8f   v; float  e[8]; };

#define WMMA_BF16(A, B, C) \
  __builtin_amdgcn_wmma_f32_16x16x32_bf16(false, (A), false, (B), (short)0, (C), false, false)

__device__ __forceinline__ float silu_f(float x) {
  return x * __frcp_rn(1.0f + __expf(-x));
}

// ---------------------------------------------------------------------------
// K0: pack W1 -> W1t (bf16, [256][512] N-major), W2 -> W2t (bf16, [16][256],
//     columns 9..15 zero).  131072 + 4096 elements.
// ---------------------------------------------------------------------------
__global__ __launch_bounds__(256) void grpv_prep_kernel(
    const float* __restrict__ W1, const float* __restrict__ W2,
    __bf16* __restrict__ W1t, __bf16* __restrict__ W2t) {
  int idx = blockIdx.x * 256 + threadIdx.x;
  if (idx < 256 * 512) {
    int n = idx >> 9;          // 0..255
    int k = idx & 511;         // 0..511
    W1t[(size_t)n * 512 + k] = (__bf16)W1[(size_t)k * 256 + n];
  } else {
    int j = idx - 256 * 512;
    if (j < 16 * 256) {
      int n = j >> 8;          // 0..15
      int k = j & 255;         // 0..255
      W2t[(size_t)n * 256 + k] = (n < 9) ? (__bf16)W2[(size_t)k * 9 + n] : (__bf16)0.0f;
    }
  }
}

// ---------------------------------------------------------------------------
// K1: fused mean + GEMM1(WMMA) + silu + GEMM2(WMMA) + rotation + transform + v
//     grid.x = (B*G)/16 = 3072 M-tiles, 256 threads (8 waves).
// ---------------------------------------------------------------------------
__global__ __launch_bounds__(256) void grpv_fused_kernel(
    const float* __restrict__ hand_tokens,   // (B,24,512)
    const __bf16* __restrict__ W1t,          // (256,512) bf16 N-major
    const float* __restrict__ b1,            // (256)
    const __bf16* __restrict__ W2t,          // (16,256) bf16 N-major padded
    const float* __restrict__ b2,            // (9)
    const float* __restrict__ keypoints,     // (B,24,3)
    const float* __restrict__ timesteps,     // (B)
    const float* __restrict__ template_xyz,  // (24,3)
    const int*   __restrict__ grp,           // (6,4)
    float* __restrict__ Vbuf)                // (B,24,3)
{
  __shared__ __bf16 As[16][512];   // averaged group tokens (A-tile), 16 KB
  __shared__ __bf16 Hs[16][256];   // silu hidden (A-tile for GEMM2), 8 KB
  __shared__ float  Ps[16][12];    // params per group-row

  const int tid  = threadIdx.x;
  const int lane = tid & 31;
  const int wave = tid >> 5;
  const int lrow = lane & 15;      // M (A) or N (B/C) within tile
  const int lh   = lane >> 4;      // lane half
  const int mrow0 = blockIdx.x * 16;

  // ---- Stage 1: streaming mean of 4 token rows -> bf16 A-tile in LDS ------
  for (int t = tid; t < 16 * 128; t += 256) {
    int r = t >> 7;                // tile row 0..15
    int c = t & 127;               // float4 chunk 0..127
    int row = mrow0 + r;
    int bb = row / 6, gg = row % 6;
    int t0 = grp[gg * 4 + 0], t1 = grp[gg * 4 + 1];
    int t2 = grp[gg * 4 + 2], t3 = grp[gg * 4 + 3];
    size_t base = (size_t)bb * 24 * 512 + (size_t)(c << 2);
    float4 s0 = *(const float4*)(hand_tokens + base + (size_t)t0 * 512);
    float4 s1 = *(const float4*)(hand_tokens + base + (size_t)t1 * 512);
    float4 s2 = *(const float4*)(hand_tokens + base + (size_t)t2 * 512);
    float4 s3 = *(const float4*)(hand_tokens + base + (size_t)t3 * 512);
    int k = c << 2;
    As[r][k + 0] = (__bf16)(0.25f * (s0.x + s1.x + s2.x + s3.x));
    As[r][k + 1] = (__bf16)(0.25f * (s0.y + s1.y + s2.y + s3.y));
    As[r][k + 2] = (__bf16)(0.25f * (s0.z + s1.z + s2.z + s3.z));
    As[r][k + 3] = (__bf16)(0.25f * (s0.w + s1.w + s2.w + s3.w));
  }
  __syncthreads();

  // ---- Stage 2: GEMM1 — each wave owns two 16-col N-tiles ------------------
  const int n0 = wave * 2;          // N-tiles 2w, 2w+1
  FragC acc0, acc1;
  acc0.v = (v8f){0.f,0.f,0.f,0.f,0.f,0.f,0.f,0.f};
  acc1.v = acc0.v;

  const int Ncol0 = n0 * 16 + lrow;
  const int Ncol1 = Ncol0 + 16;

#pragma unroll 4
  for (int kc = 0; kc < 16; ++kc) {
    const int K0 = kc * 32;
    // A fragment: contiguous K layout  K = lh*16 + 2j + p
    FragAB a;
#pragma unroll
    for (int j = 0; j < 8; ++j) {
      int kk = K0 + lh * 16 + 2 * j;
      a.e[2 * j + 0] = As[lrow][kk + 0];
      a.e[2 * j + 1] = As[lrow][kk + 1];
    }
    // B fragments: per lane 16 contiguous K bf16 (32 B) from N-major W1t
    FragAB b0, b1f;
    const __bf16* bp0 = W1t + (size_t)Ncol0 * 512 + K0 + lh * 16;
    const __bf16* bp1 = W1t + (size_t)Ncol1 * 512 + K0 + lh * 16;
    b0.q[0]  = *(const uint4*)(bp0);
    b0.q[1]  = *(const uint4*)(bp0 + 8);
    b1f.q[0] = *(const uint4*)(bp1);
    b1f.q[1] = *(const uint4*)(bp1 + 8);

    acc0.v = WMMA_BF16(a.v, b0.v,  acc0.v);
    acc1.v = WMMA_BF16(a.v, b1f.v, acc1.v);
  }

  // ---- Stage 3: +b1, silu, store hidden tile as bf16 ----------------------
  {
    float bb0 = b1[Ncol0];
    float bb1 = b1[Ncol1];
#pragma unroll
    for (int i = 0; i < 8; ++i) {
      int m = i + lh * 8;          // C/D layout: M = i + (lane/16)*8, N = lane%16
      Hs[m][Ncol0] = (__bf16)silu_f(acc0.e[i] + bb0);
      Hs[m][Ncol1] = (__bf16)silu_f(acc1.e[i] + bb1);
    }
  }
  __syncthreads();

  // ---- Stage 4: GEMM2 (16x256 @ 256x16pad), wave 0 only -------------------
  if (wave == 0) {
    FragC acc2;
    acc2.v = (v8f){0.f,0.f,0.f,0.f,0.f,0.f,0.f,0.f};
#pragma unroll
    for (int kc = 0; kc < 8; ++kc) {
      const int K0 = kc * 32;
      FragAB a;
#pragma unroll
      for (int j = 0; j < 8; ++j) {
        int kk = K0 + lh * 16 + 2 * j;
        a.e[2 * j + 0] = Hs[lrow][kk + 0];
        a.e[2 * j + 1] = Hs[lrow][kk + 1];
      }
      FragAB b;
      const __bf16* bp = W2t + (size_t)lrow * 256 + K0 + lh * 16;
      b.q[0] = *(const uint4*)(bp);
      b.q[1] = *(const uint4*)(bp + 8);
      acc2.v = WMMA_BF16(a.v, b.v, acc2.v);
    }
    if (lrow < 9) {
      float bb = b2[lrow];
#pragma unroll
      for (int i = 0; i < 8; ++i)
        Ps[i + lh * 8][lrow] = acc2.e[i] + bb;
    }
  }
  __syncthreads();

  // ---- Stage 5: 6D rotation, rigid transform, v = (x_goal - kp)/tau -------
  if (tid < 64) {
    int r = tid >> 2, s = tid & 3;
    int row = mrow0 + r;
    int bb = row / 6, gg = row % 6;
    int node = grp[gg * 4 + s];

    float p0 = Ps[r][0], p1 = Ps[r][1], p2 = Ps[r][2];
    float p3 = Ps[r][3], p4 = Ps[r][4], p5 = Ps[r][5];
    float t0 = Ps[r][6], t1 = Ps[r][7], t2 = Ps[r][8];

    // x = normalize(p0:3)
    float nx = sqrtf(p0 * p0 + p1 * p1 + p2 * p2);
    float inx = 1.0f / fmaxf(nx, 1e-8f);
    float xx = p0 * inx, xy = p1 * inx, xz = p2 * inx;
    // z = normalize(cross(x, p3:6))
    float cx = xy * p5 - xz * p4;
    float cy = xz * p3 - xx * p5;
    float cz = xx * p4 - xy * p3;
    float nz = sqrtf(cx * cx + cy * cy + cz * cz);
    float inz = 1.0f / fmaxf(nz, 1e-8f);
    float zx = cx * inz, zy = cy * inz, zz = cz * inz;
    // y = cross(z, x)
    float yx = zy * xz - zz * xy;
    float yy = zz * xx - zx * xz;
    float yz = zx * xy - zy * xx;

    const float* tp = template_xyz + (size_t)node * 3;
    float px = tp[0], py = tp[1], pz = tp[2];
    // R columns are (x,y,z): R*p = x*px + y*py + z*pz
    float gx = xx * px + yx * py + zx * pz + t0;
    float gy = xy * px + yy * py + zy * pz + t1;
    float gz = xz * px + yz * py + zz * pz + t2;

    float tau = fmaxf(1.0f - timesteps[bb], 0.001f);
    float itau = 1.0f / tau;
    size_t o = ((size_t)bb * 24 + node) * 3;
    Vbuf[o + 0] = (gx - keypoints[o + 0]) * itau;
    Vbuf[o + 1] = (gy - keypoints[o + 1]) * itau;
    Vbuf[o + 2] = (gz - keypoints[o + 2]) * itau;
  }
}

// ---------------------------------------------------------------------------
// K2: tangent projection. The chain edge set makes the 23x23 system symmetric
//     tridiagonal: diag = 8|dx_e|^2 + eps, off = -4 dx_e . dx_{e+1}.
//     One 32-lane group per batch element; Thomas solve by lane 0.
//     grid.x = B/8 = 1024, 256 threads (8 batch elements per block).
// ---------------------------------------------------------------------------
#define NEDGE 23
#define EPSV 1e-6f

__global__ __launch_bounds__(256) void grpv_tangent_kernel(
    const float* __restrict__ keypoints,   // (B,24,3)
    const float* __restrict__ Vbuf,        // (B,24,3)
    float* __restrict__ out)               // (B,24,3)
{
  __shared__ float dxs[8][NEDGE][3];
  __shared__ float diag[8][NEDGE];
  __shared__ float off[8][NEDGE];
  __shared__ float rhs[8][NEDGE];
  __shared__ float lam[8][NEDGE];
  __shared__ float cp[8][NEDGE];

  const int tid  = threadIdx.x;
  const int sub  = tid >> 5;     // 0..7: batch slot
  const int lane = tid & 31;
  const int b    = blockIdx.x * 8 + sub;
  const size_t base = (size_t)b * 24 * 3;

  if (lane < NEDGE) {
    int e = lane;
    float x0 = keypoints[base + e * 3 + 0], x1 = keypoints[base + e * 3 + 1], x2 = keypoints[base + e * 3 + 2];
    float w0 = keypoints[base + (e + 1) * 3 + 0], w1 = keypoints[base + (e + 1) * 3 + 1], w2 = keypoints[base + (e + 1) * 3 + 2];
    float v0 = Vbuf[base + e * 3 + 0], v1 = Vbuf[base + e * 3 + 1], v2 = Vbuf[base + e * 3 + 2];
    float u0 = Vbuf[base + (e + 1) * 3 + 0], u1 = Vbuf[base + (e + 1) * 3 + 1], u2 = Vbuf[base + (e + 1) * 3 + 2];
    float dx0 = x0 - w0, dx1 = x1 - w1, dx2 = x2 - w2;
    float dv0 = v0 - u0, dv1 = v1 - u1, dv2 = v2 - u2;
    dxs[sub][e][0] = dx0; dxs[sub][e][1] = dx1; dxs[sub][e][2] = dx2;
    rhs[sub][e]  = 2.0f * (dx0 * dv0 + dx1 * dv1 + dx2 * dv2);
    diag[sub][e] = 8.0f * (dx0 * dx0 + dx1 * dx1 + dx2 * dx2) + EPSV;
  }
  __syncthreads();
  if (lane < NEDGE - 1) {
    int e = lane;
    off[sub][e] = -4.0f * (dxs[sub][e][0] * dxs[sub][e + 1][0] +
                           dxs[sub][e][1] * dxs[sub][e + 1][1] +
                           dxs[sub][e][2] * dxs[sub][e + 1][2]);
  }
  __syncthreads();

  if (lane == 0) {
    // Thomas algorithm (symmetric tridiagonal)
    float m = diag[sub][0];
    cp[sub][0]  = off[sub][0] / m;
    lam[sub][0] = rhs[sub][0] / m;    // lam holds d' during forward sweep
    for (int e = 1; e < NEDGE; ++e) {
      float sd = off[sub][e - 1];
      m = diag[sub][e] - sd * cp[sub][e - 1];
      float im = 1.0f / m;
      cp[sub][e]  = (e < NEDGE - 1) ? off[sub][e] * im : 0.0f;
      lam[sub][e] = (rhs[sub][e] - sd * lam[sub][e - 1]) * im;
    }
    for (int e = NEDGE - 2; e >= 0; --e)
      lam[sub][e] -= cp[sub][e] * lam[sub][e + 1];
  }
  __syncthreads();

  if (lane < 24) {
    int n = lane;
    float c0 = 0.f, c1 = 0.f, c2 = 0.f;
    if (n < NEDGE) {
      float l = 2.0f * lam[sub][n];
      c0 += l * dxs[sub][n][0]; c1 += l * dxs[sub][n][1]; c2 += l * dxs[sub][n][2];
    }
    if (n >= 1) {
      float l = 2.0f * lam[sub][n - 1];
      c0 -= l * dxs[sub][n - 1][0]; c1 -= l * dxs[sub][n - 1][1]; c2 -= l * dxs[sub][n - 1][2];
    }
    out[base + n * 3 + 0] = Vbuf[base + n * 3 + 0] - c0;
    out[base + n * 3 + 1] = Vbuf[base + n * 3 + 1] - c1;
    out[base + n * 3 + 2] = Vbuf[base + n * 3 + 2] - c2;
  }
}

// ---------------------------------------------------------------------------
extern "C" void kernel_launch(void* const* d_in, const int* in_sizes, int n_in,
                              void* d_out, int out_size, void* d_ws, size_t ws_size,
                              hipStream_t stream) {
  const float* keypoints    = (const float*)d_in[0];
  const float* timesteps    = (const float*)d_in[1];
  const float* hand_tokens  = (const float*)d_in[2];
  const float* W1           = (const float*)d_in[3];
  const float* b1           = (const float*)d_in[4];
  const float* W2           = (const float*)d_in[5];
  const float* b2           = (const float*)d_in[6];
  const float* template_xyz = (const float*)d_in[7];
  const int*   groups       = (const int*)d_in[8];
  float* out = (float*)d_out;

  // workspace layout
  char* ws = (char*)d_ws;
  __bf16* W1t = (__bf16*)ws;                       // 256*512*2 = 262144 B
  __bf16* W2t = (__bf16*)(ws + 262144);            // 16*256*2  = 8192 B
  float*  Vbuf = (float*)(ws + 262144 + 8192);     // 8192*24*3*4 = 2359296 B

  const int B = 8192, G = 6;

  // K0: weight packing (bf16, N-major)
  grpv_prep_kernel<<<(256 * 512 + 16 * 256 + 255) / 256, 256, 0, stream>>>(W1, W2, W1t, W2t);

  // K1: fused mean + WMMA GEMMs + rigid transform -> v
  grpv_fused_kernel<<<(B * G) / 16, 256, 0, stream>>>(
      hand_tokens, W1t, b1, W2t, b2, keypoints, timesteps, template_xyz, groups, Vbuf);

  // K2: tridiagonal tangent projection -> out
  grpv_tangent_kernel<<<B / 8, 256, 0, stream>>>(keypoints, Vbuf, out);
}
